// HypergraphTransformer_24008867185108
// MI455X (gfx1250) — matrix operand order repaired and checked by reading, not compiled, not run on hardware
//
#include <hip/hip_runtime.h>
#include <hip/hip_bf16.h>
#include <math.h>

typedef __attribute__((ext_vector_type(16))) _Float16 v16h;
typedef __attribute__((ext_vector_type(8)))  _Float16 v8h;
typedef __attribute__((ext_vector_type(4)))  _Float16 v4h;
typedef __attribute__((ext_vector_type(2)))  _Float16 h2;
typedef __attribute__((ext_vector_type(8)))  float    v8f;
typedef __attribute__((ext_vector_type(4)))  int      v4i;

static constexpr int B_    = 256;
static constexpr int NQ    = 30;
static constexpr int NQP   = 32;    // padded question nodes
static constexpr int NK    = 100;
static constexpr int NKP   = 128;   // padded kg nodes (mult of 32 for K-dim use)
static constexpr int HID_  = 512;
static constexpr int K0    = 900;   // WORD_EMB * 3
static constexpr int K0P   = 928;   // padded to mult of 32
static constexpr int NOUT  = 300;
static constexpr int NOUTP = 320;   // padded
static constexpr int NANS  = 32000;
static constexpr int DIMF  = 1024;  // final concat dim

// CDNA5 async global->LDS copy path (guarded: falls back to plain copies).
#if defined(__has_builtin)
#  if __has_builtin(__builtin_amdgcn_global_load_async_to_lds_b128) && \
      __has_builtin(__builtin_amdgcn_s_wait_asynccnt)
#    define USE_ASYNC_LDS 1
#  endif
#endif
#ifndef USE_ASYNC_LDS
#  define USE_ASYNC_LDS 0
#endif

#if USE_ASYNC_LDS
typedef __attribute__((address_space(1))) v4i gas_v4i;  // global int4
typedef __attribute__((address_space(3))) v4i las_v4i;  // LDS int4
#endif

__device__ __forceinline__ void g2l_16B(_Float16* l, const _Float16* g) {
#if USE_ASYNC_LDS
  __builtin_amdgcn_global_load_async_to_lds_b128((gas_v4i*)g, (las_v4i*)l, 0, 0);
#else
  *(v8h*)l = *(const v8h*)g;
#endif
}
__device__ __forceinline__ void async_fence() {
#if USE_ASYNC_LDS
  __builtin_amdgcn_s_wait_asynccnt(0);
#endif
}

__device__ __forceinline__ float wave_sum(float v) {
  #pragma unroll
  for (int o = 16; o > 0; o >>= 1) v += __shfl_down(v, o, 32);
  return __shfl(v, 0, 32);
}
__device__ __forceinline__ float wave_max(float v) {
  #pragma unroll
  for (int o = 16; o > 0; o >>= 1) v = fmaxf(v, __shfl_down(v, o, 32));
  return __shfl(v, 0, 32);
}

// ---------------------------------------------------------------------------
// LDS-staged WMMA GEMM: C = act(alpha * A(MxK) * B(KxN) + bias + Dres)
// Block = 8 waves = 16(M) x 128(N) slab, K step 32, double-buffered LDS.
// A tile (16x32) shared by all waves; B tile staged TRANSPOSED ([n][k]) so
// fragment reads are contiguous and match the ISA 7.12.2 register layouts.
// BT=true: B stored N x K row-major (straight copy -> async-to-LDS path).
// All dims pre-padded: M,N mult of 16, K mult of 32.
// ---------------------------------------------------------------------------
template <bool BT>
__global__ __launch_bounds__(256)
void gemm16(const _Float16* __restrict__ A, long sA, int lda,
            const _Float16* __restrict__ Bm, long sB, int ldb,
            const float* __restrict__ Dres, long sD, int ldd,
            float* __restrict__ C, long sC, int ldc,
            _Float16* __restrict__ C16, long sC16, int ldc16,
            int M, int N, int K,
            const float* __restrict__ bias, int bias_len,
            int act, float alpha) {
  __shared__ __align__(16) _Float16 As[2][16][48];    // row-major [m][k]
  __shared__ __align__(16) _Float16 Bs[2][128][48];   // transposed [n][k]
  const int tid  = threadIdx.x;
  const int lane = tid & 31;
  const int wave = tid >> 5;
  const int m0   = blockIdx.x * 16;
  const int nblk = blockIdx.y * 128;
  const int n0   = nblk + wave * 16;
  const int b    = blockIdx.z;
  const bool active = (m0 < M) && (n0 < N);   // wave-uniform; no early return
  const _Float16* Ab = A + (long)b * sA;
  const _Float16* Bb = Bm + (long)b * sB;
  const int idx16 = lane & 15;
  const int half  = lane >> 4;

  auto stage = [&](int buf, int k0) {
    {  // A tile: 16 rows x 32 cols, 2 f16 per thread
      const int r = tid >> 4;
      const int c = (tid & 15) * 2;
      *(h2*)&As[buf][r][c] = *(const h2*)(Ab + (long)(m0 + r) * lda + k0 + c);
    }
    if constexpr (BT) {  // straight 32B row copies -> async path
      const int n = tid >> 1;
      const int c = (tid & 1) * 16;
      const _Float16* src = Bb + (long)(nblk + n) * ldb + k0 + c;
      g2l_16B(&Bs[buf][n][c], src);
      g2l_16B(&Bs[buf][n][c + 8], src + 8);
    } else {             // K x N source: transpose through LDS
      const int kk = tid >> 3;
      const int c  = (tid & 7) * 16;
      const _Float16* src = Bb + (long)(k0 + kk) * ldb + nblk + c;
      v8h x0 = *(const v8h*)src;
      v8h x1 = *(const v8h*)(src + 8);
      #pragma unroll
      for (int i = 0; i < 8; ++i) {
        Bs[buf][c + i][kk]     = x0[i];
        Bs[buf][c + 8 + i][kk] = x1[i];
      }
    }
  };

  v8f c = {};
  stage(0, 0);
  async_fence();
  __syncthreads();
  const int nIter = K / 32;
  for (int it = 0; it < nIter; ++it) {
    const int cur = it & 1;
    if (it + 1 < nIter) stage(cur ^ 1, (it + 1) * 32);
    if (active) {
      const _Float16* arow = &As[cur][idx16][0];
      const _Float16* brow = &Bs[cur][wave * 16 + idx16][0];
      v16h av, bv;
      v4h t;
      t = *(const v4h*)(arow + 8 * half);
      #pragma unroll
      for (int i = 0; i < 4; ++i) av[i] = t[i];
      t = *(const v4h*)(arow + 8 * half + 4);
      #pragma unroll
      for (int i = 0; i < 4; ++i) av[4 + i] = t[i];
      t = *(const v4h*)(arow + 16 + 8 * half);
      #pragma unroll
      for (int i = 0; i < 4; ++i) av[8 + i] = t[i];
      t = *(const v4h*)(arow + 20 + 8 * half);
      #pragma unroll
      for (int i = 0; i < 4; ++i) av[12 + i] = t[i];
      t = *(const v4h*)(brow + 8 * half);
      #pragma unroll
      for (int i = 0; i < 4; ++i) bv[i] = t[i];
      t = *(const v4h*)(brow + 8 * half + 4);
      #pragma unroll
      for (int i = 0; i < 4; ++i) bv[4 + i] = t[i];
      t = *(const v4h*)(brow + 16 + 8 * half);
      #pragma unroll
      for (int i = 0; i < 4; ++i) bv[8 + i] = t[i];
      t = *(const v4h*)(brow + 20 + 8 * half);
      #pragma unroll
      for (int i = 0; i < 4; ++i) bv[12 + i] = t[i];
      c = __builtin_amdgcn_wmma_f32_16x16x32_f16(false, av, false, bv,
                                                 (short)0, c, false, false);
    }
    async_fence();
    __syncthreads();
  }

  if (active) {
    // C layout: lane = col(idx16) + 16*half ; vgpr r -> row = r + 8*half
    #pragma unroll
    for (int r = 0; r < 8; ++r) {
      const int row = m0 + r + 8 * half;
      const int col = n0 + idx16;
      float v = c[r] * alpha;
      if (bias) v += (col < bias_len) ? bias[col] : 0.f;
      if (Dres) v += Dres[(long)b * sD + (long)row * ldd + col];
      if (act == 1) v = fmaxf(v, 0.f);
      if (C)   C  [(long)b * sC   + (long)row * ldc   + col] = v;
      if (C16) C16[(long)b * sC16 + (long)row * ldc16 + col] = (_Float16)v;
    }
  }
}

// Gather word embeddings, NaN->0, write padded f16 activation matrix.
__global__ __launch_bounds__(256)
void gather_embed(const int* __restrict__ ids, const float* __restrict__ emb,
                  _Float16* __restrict__ dst, int realNodes, int padNodes,
                  long total) {
  long idx = (long)blockIdx.x * 256 + threadIdx.x;
  if (idx >= total) return;
  const int  kcol = (int)(idx % K0P);
  const long row  = idx / K0P;
  const int  node = (int)(row % padNodes);
  const long b    = row / padNodes;
  float val = 0.f;
  if (node < realNodes && kcol < K0) {
    const int word = kcol / 300;
    const int off  = kcol - word * 300;
    const int id   = ids[(b * realNodes + node) * 3 + word];
    const float v  = emb[(long)id * 300 + off];
    val = (v == v) ? v : 0.f;
  }
  dst[idx] = (_Float16)val;
}

// fp32 -> f16 with zero padding (dst R2 x C2 >= src R x C).
__global__ __launch_bounds__(256)
void convert_pad(const float* __restrict__ src, int R, int C,
                 _Float16* __restrict__ dst, int R2, int C2) {
  long idx = (long)blockIdx.x * 256 + threadIdx.x;
  const long total = (long)R2 * C2;
  if (idx >= total) return;
  const int  c = (int)(idx % C2);
  const long r = idx / C2;
  const float v = (r < R && c < C) ? src[r * C + c] : 0.f;
  dst[idx] = (_Float16)v;
}

// y = proj(expmap0(proj_tan0(x)), c), dim=512, one wave per row (in-place ok).
__global__ __launch_bounds__(256)
void hyper_emb(const float* __restrict__ x, float* __restrict__ y, long rows,
               const float* __restrict__ curv, int ci) {
  const long row = (long)blockIdx.x * 8 + (threadIdx.x >> 5);
  if (row >= rows) return;
  const int lane = threadIdx.x & 31;
  const float c = curv[ci];
  const float Kc = 1.f / c;
  const float sqrtK = sqrtf(Kc);
  const float* xr = x + row * (long)HID_;
  float* yr = y + row * (long)HID_;
  float d[16];
  float ss = 0.f;
  #pragma unroll
  for (int i = 0; i < 16; ++i) {
    const int j = lane + 32 * i;
    const float v = (j == 0) ? 0.f : xr[j];   // proj_tan0
    d[i] = v;
    ss += v * v;
  }
  ss = wave_sum(ss);
  const float n = fmaxf(sqrtf(ss), 1e-15f);
  const float theta = n / sqrtK;
  const float sh = sinhf(theta);
  const float coef = sqrtK * sh / n;
  #pragma unroll
  for (int i = 0; i < 16; ++i) {
    const int j = lane + 32 * i;
    if (j) yr[j] = coef * d[i];
  }
  if (lane == 0) yr[0] = sqrtf(fmaxf(Kc + Kc * sh * sh, 1e-7f)); // proj()
}

// u = proj_tan0(logmap0(x, c)), generic dim, fp32 + optional padded f16 out.
__global__ __launch_bounds__(256)
void logmap_tan(const float* __restrict__ x, float* __restrict__ u,
                _Float16* __restrict__ u16, long rows, int dim,
                const float* __restrict__ curv, int ci) {
  const long row = (long)blockIdx.x * 8 + (threadIdx.x >> 5);
  if (row >= rows) return;
  const int lane = threadIdx.x & 31;
  const float c = curv[ci];
  const float Kc = 1.f / c;
  const float sqrtK = sqrtf(Kc);
  const float* xr = x + row * (long)dim;
  float ss = 0.f;
  for (int j = lane; j < dim; j += 32)
    if (j) { const float v = xr[j]; ss += v * v; }
  ss = wave_sum(ss);
  const float n  = fmaxf(sqrtf(ss), 1e-15f);
  const float t  = fmaxf(xr[0] / sqrtK, 1.f + 1e-7f);
  const float coef = sqrtK * acoshf(t) / n;
  float* ur = u + row * (long)dim;
  _Float16* hr = u16 ? (u16 + row * (long)dim) : nullptr;
  for (int j = lane; j < dim; j += 32) {
    const float v = j ? coef * xr[j] : 0.f;
    ur[j] = v;
    if (hr) hr[j] = (_Float16)v;
  }
}

// softmax over last dim (32 padded, realCols real). Padded rows/cols -> 0.
__global__ __launch_bounds__(256)
void softmax_lastdim(const float* __restrict__ s, _Float16* __restrict__ p,
                     long totalRows, int rowsPerB, int realRows, int realCols) {
  const long row = (long)blockIdx.x * 8 + (threadIdx.x >> 5);
  if (row >= totalRows) return;
  const int lane = threadIdx.x & 31;
  const int r = (int)(row % rowsPerB);
  _Float16* pr = p + row * 32;
  if (r >= realRows) { pr[lane] = (_Float16)0.f; return; }  // wave-uniform
  const float* sr = s + row * 32;
  const float v = (lane < realCols) ? sr[lane] : -3.4e38f;
  const float m = wave_max(v);
  const float e = (lane < realCols) ? expf(v - m) : 0.f;
  const float sum = wave_sum(e);
  pr[lane] = (_Float16)(e / sum);
}

// softmax over k axis of s (B,NKP,NQP): pq[b,q,k] with k-padding -> 0.
__global__ __launch_bounds__(256)
void softmax_kaxis(const float* __restrict__ s, _Float16* __restrict__ p,
                   long totalRows) {
  const long row = (long)blockIdx.x * 8 + (threadIdx.x >> 5);
  if (row >= totalRows) return;
  const int lane = threadIdx.x & 31;
  const long b = row / NQP;
  const int  q = (int)(row % NQP);
  const float* sb = s + b * (long)(NKP * NQP);
  float vals[4];
  float m = -3.4e38f;
  #pragma unroll
  for (int i = 0; i < 4; ++i) {
    const int k = lane + 32 * i;
    const float v = (k < NK) ? sb[(long)k * NQP + q] : -3.4e38f;
    vals[i] = v;
    m = fmaxf(m, v);
  }
  m = wave_max(m);
  float e[4];
  float sum = 0.f;
  #pragma unroll
  for (int i = 0; i < 4; ++i) {
    const int k = lane + 32 * i;
    e[i] = (k < NK) ? expf(vals[i] - m) : 0.f;
    sum += e[i];
  }
  sum = wave_sum(sum);
  const float inv = 1.f / sum;
  _Float16* pr = p + row * (long)NKP;
  #pragma unroll
  for (int i = 0; i < 4; ++i) pr[lane + 32 * i] = (_Float16)(e[i] * inv);
}

// last_kq = concat(mean_k over 100 real rows, mean_q over 30 real rows).
__global__ __launch_bounds__(256)
void mean_concat(const float* __restrict__ k, const float* __restrict__ q,
                 float* __restrict__ outv) {
  const long idx = (long)blockIdx.x * 256 + threadIdx.x;  // B*1024
  const int  j = (int)(idx % DIMF);
  const long b = idx / DIMF;
  float s = 0.f;
  if (j < HID_) {
    const float* kr = k + (b * NKP) * (long)HID_ + j;
    for (int r = 0; r < NK; ++r) s += kr[(long)r * HID_];
    s *= (1.f / NK);
  } else {
    const float* qr = q + (b * NQP) * (long)HID_ + (j - HID_);
    for (int r = 0; r < NQ; ++r) s += qr[(long)r * HID_];
    s *= (1.f / NQ);
  }
  outv[idx] = s;
}

// In-place row log-softmax, one block per row.
__global__ __launch_bounds__(256)
void log_softmax_rows(float* __restrict__ x, int cols) {
  __shared__ float red[256];
  float* row = x + (long)blockIdx.x * cols;
  const int t = threadIdx.x;
  float m = -3.4e38f;
  for (int j = t; j < cols; j += 256) m = fmaxf(m, row[j]);
  red[t] = m; __syncthreads();
  for (int s2 = 128; s2 > 0; s2 >>= 1) {
    if (t < s2) red[t] = fmaxf(red[t], red[t + s2]);
    __syncthreads();
  }
  m = red[0]; __syncthreads();
  float s = 0.f;
  for (int j = t; j < cols; j += 256) s += expf(row[j] - m);
  red[t] = s; __syncthreads();
  for (int s2 = 128; s2 > 0; s2 >>= 1) {
    if (t < s2) red[t] += red[t + s2];
    __syncthreads();
  }
  const float lg = m + logf(red[0]);
  for (int j = t; j < cols; j += 256) row[j] = row[j] - lg;
}

static inline unsigned cdiv(long a, long b) { return (unsigned)((a + b - 1) / b); }

extern "C" void kernel_launch(void* const* d_in, const int* in_sizes, int n_in,
                              void* d_out, int out_size, void* d_ws, size_t ws_size,
                              hipStream_t stream) {
  (void)in_sizes; (void)n_in; (void)out_size; (void)ws_size;
  const int*   heq_id = (const int*)d_in[0];
  const int*   hek_id = (const int*)d_in[1];
  const float* emb    = (const float*)d_in[2];
  const float* q2h_w  = (const float*)d_in[3];
  const float* q2h_b  = (const float*)d_in[4];
  const float* k2h_w  = (const float*)d_in[5];
  const float* k2h_b  = (const float*)d_in[6];
  const float* Wk     = (const float*)d_in[7];
  const float* Wq     = (const float*)d_in[8];
  const float* curv   = (const float*)d_in[9];
  const float* p1w    = (const float*)d_in[10];
  const float* p1b    = (const float*)d_in[11];
  const float* p2w    = (const float*)d_in[12];
  const float* p2b    = (const float*)d_in[13];
  const float* glove  = (const float*)d_in[14];
  float* outp = (float*)d_out;

  char* base = (char*)d_ws;
  size_t off = 0;
  auto alloc = [&](size_t bytes) -> char* {
    const size_t a = (off + 255) & ~(size_t)255;
    off = a + bytes;
    return base + a;
  };

  // Shared arena: gather buffers (dead after input GEMMs) vs attention outputs.
  const size_t heq_sz  = (size_t)B_ * NQP * K0P * 2;
  const size_t hek_sz  = (size_t)B_ * NKP * K0P * 2;
  const size_t attk_sz = (size_t)B_ * NKP * HID_ * 4;
  const size_t attq_sz = (size_t)B_ * NQP * HID_ * 4;
  const size_t pairA = ((heq_sz + 255) & ~(size_t)255) + hek_sz;
  const size_t pairB = ((attk_sz + 255) & ~(size_t)255) + attq_sz;
  char* arena = alloc(pairA > pairB ? pairA : pairB);
  _Float16* heq16 = (_Float16*)arena;
  _Float16* hek16 = (_Float16*)(arena + ((heq_sz + 255) & ~(size_t)255));
  float* attk = (float*)arena;
  float* attq = (float*)(arena + ((attk_sz + 255) & ~(size_t)255));

  _Float16* q2h16 = (_Float16*)alloc((size_t)K0P * HID_ * 2);
  _Float16* k2h16 = (_Float16*)alloc((size_t)K0P * HID_ * 2);
  _Float16* Wk16  = (_Float16*)alloc((size_t)3 * HID_ * HID_ * 2);
  _Float16* Wq16  = (_Float16*)alloc((size_t)3 * HID_ * HID_ * 2);
  _Float16* p116  = (_Float16*)alloc((size_t)DIMF * HID_ * 2);
  _Float16* p216  = (_Float16*)alloc((size_t)HID_ * NOUTP * 2);
  _Float16* gl16  = (_Float16*)alloc((size_t)NANS * NOUTP * 2);
  float*    kcur  = (float*)alloc((size_t)B_ * NKP * HID_ * 4);
  float*    qcur  = (float*)alloc((size_t)B_ * NQP * HID_ * 4);
  float*    kt    = (float*)alloc((size_t)B_ * NKP * HID_ * 4);
  float*    qt    = (float*)alloc((size_t)B_ * NQP * HID_ * 4);
  _Float16* kt16  = (_Float16*)alloc((size_t)B_ * NKP * HID_ * 2);
  _Float16* qt16  = (_Float16*)alloc((size_t)B_ * NQP * HID_ * 2);
  _Float16* kp16  = (_Float16*)alloc((size_t)B_ * NKP * HID_ * 2);
  _Float16* qp16  = (_Float16*)alloc((size_t)B_ * NQP * HID_ * 2);
  float*    sbuf  = (float*)alloc((size_t)B_ * NKP * NQP * 4);
  _Float16* pk16  = (_Float16*)alloc((size_t)B_ * NKP * NQP * 2);
  _Float16* pq16  = (_Float16*)alloc((size_t)B_ * NQP * NKP * 2);
  float*    lkq   = (float*)alloc((size_t)B_ * DIMF * 4);
  _Float16* euc16 = (_Float16*)alloc((size_t)B_ * DIMF * 2);
  _Float16* h16   = (_Float16*)alloc((size_t)B_ * HID_ * 2);
  _Float16* out16 = (_Float16*)alloc((size_t)B_ * NOUTP * 2);

  // --- weight conversion (f32 -> padded f16) ---
  auto cvt = [&](const float* s, int R, int C, _Float16* d, int R2, int C2) {
    const long n = (long)R2 * C2;
    convert_pad<<<cdiv(n, 256), 256, 0, stream>>>(s, R, C, d, R2, C2);
  };
  cvt(q2h_w, K0, HID_, q2h16, K0P, HID_);
  cvt(k2h_w, K0, HID_, k2h16, K0P, HID_);
  cvt(Wk, 3 * HID_, HID_, Wk16, 3 * HID_, HID_);
  cvt(Wq, 3 * HID_, HID_, Wq16, 3 * HID_, HID_);
  cvt(p1w, DIMF, HID_, p116, DIMF, HID_);
  cvt(p2w, HID_, NOUT, p216, HID_, NOUTP);
  cvt(glove, NANS, NOUT, gl16, NANS, NOUTP);

  // --- gather + NaN clean ---
  gather_embed<<<cdiv((long)B_ * NQP * K0P, 256), 256, 0, stream>>>(
      heq_id, emb, heq16, NQ, NQP, (long)B_ * NQP * K0P);
  gather_embed<<<cdiv((long)B_ * NKP * K0P, 256), 256, 0, stream>>>(
      hek_id, emb, hek16, NK, NKP, (long)B_ * NKP * K0P);

  // --- input projections: (B*Npad, 928) x (928, 512) + bias ---
  gemm16<false><<<dim3(B_ * NQP / 16, HID_ / 128, 1), 256, 0, stream>>>(
      heq16, 0, K0P, q2h16, 0, HID_, nullptr, 0, 0,
      qcur, 0, HID_, nullptr, 0, 0,
      B_ * NQP, HID_, K0P, q2h_b, HID_, 0, 1.f);
  gemm16<false><<<dim3(B_ * NKP / 16, HID_ / 128, 1), 256, 0, stream>>>(
      hek16, 0, K0P, k2h16, 0, HID_, nullptr, 0, 0,
      kcur, 0, HID_, nullptr, 0, 0,
      B_ * NKP, HID_, K0P, k2h_b, HID_, 0, 1.f);

  // --- lift to hyperboloid with c0 ---
  hyper_emb<<<cdiv((long)B_ * NQP, 8), 256, 0, stream>>>(qcur, qcur, (long)B_ * NQP, curv, 0);
  hyper_emb<<<cdiv((long)B_ * NKP, 8), 256, 0, stream>>>(kcur, kcur, (long)B_ * NKP, curv, 0);

  const float scale = 0.044194173824159223f;  // 1/sqrt(512)
  for (int i = 0; i < 3; ++i) {
    logmap_tan<<<cdiv((long)B_ * NKP, 8), 256, 0, stream>>>(
        kcur, kt, kt16, (long)B_ * NKP, HID_, curv, i);
    logmap_tan<<<cdiv((long)B_ * NQP, 8), 256, 0, stream>>>(
        qcur, qt, qt16, (long)B_ * NQP, HID_, curv, i);
    // kp = kt @ Wk[i], qp = qt @ Wq[i]  (f16 outputs only)
    gemm16<false><<<dim3(NKP / 16, HID_ / 128, B_), 256, 0, stream>>>(
        kt16, (long)NKP * HID_, HID_, Wk16 + (size_t)i * HID_ * HID_, 0, HID_,
        nullptr, 0, 0, nullptr, 0, 0, kp16, (long)NKP * HID_, HID_,
        NKP, HID_, HID_, nullptr, 0, 0, 1.f);
    gemm16<false><<<dim3(NQP / 16, HID_ / 128, B_), 256, 0, stream>>>(
        qt16, (long)NQP * HID_, HID_, Wq16 + (size_t)i * HID_ * HID_, 0, HID_,
        nullptr, 0, 0, nullptr, 0, 0, qp16, (long)NQP * HID_, HID_,
        NQP, HID_, HID_, nullptr, 0, 0, 1.f);
    // s[b,k,q] = scale * kp . qp  (B^T layout -> async staging path)
    gemm16<true><<<dim3(NKP / 16, 1, B_), 256, 0, stream>>>(
        kp16, (long)NKP * HID_, HID_, qp16, (long)NQP * HID_, HID_,
        nullptr, 0, 0, sbuf, (long)NKP * NQP, NQP, nullptr, 0, 0,
        NKP, NQP, HID_, nullptr, 0, 0, scale);
    softmax_lastdim<<<cdiv((long)B_ * NKP, 8), 256, 0, stream>>>(
        sbuf, pk16, (long)B_ * NKP, NKP, NK, NQ);
    softmax_kaxis<<<cdiv((long)B_ * NQP, 8), 256, 0, stream>>>(
        sbuf, pq16, (long)B_ * NQP);
    // att_k = Pk @ qt + kt ; att_q = Pq @ kt + qt
    gemm16<false><<<dim3(NKP / 16, HID_ / 128, B_), 256, 0, stream>>>(
        pk16, (long)NKP * NQP, NQP, qt16, (long)NQP * HID_, HID_,
        kt, (long)NKP * HID_, HID_, attk, (long)NKP * HID_, HID_,
        nullptr, 0, 0, NKP, HID_, NQP, nullptr, 0, 0, 1.f);
    gemm16<false><<<dim3(NQP / 16, HID_ / 128, B_), 256, 0, stream>>>(
        pq16, (long)NQP * NKP, NKP, kt16, (long)NKP * HID_, HID_,
        qt, (long)NQP * HID_, HID_, attq, (long)NQP * HID_, HID_,
        nullptr, 0, 0, NQP, HID_, NKP, nullptr, 0, 0, 1.f);
    // back to hyperboloid with c_{i+1}
    hyper_emb<<<cdiv((long)B_ * NKP, 8), 256, 0, stream>>>(attk, kcur, (long)B_ * NKP, curv, i + 1);
    hyper_emb<<<cdiv((long)B_ * NQP, 8), 256, 0, stream>>>(attq, qcur, (long)B_ * NQP, curv, i + 1);
  }

  // --- head ---
  mean_concat<<<cdiv((long)B_ * DIMF, 256), 256, 0, stream>>>(kcur, qcur, lkq);
  logmap_tan<<<cdiv((long)B_, 8), 256, 0, stream>>>(lkq, lkq, euc16, (long)B_, DIMF, curv, 3);
  gemm16<false><<<dim3(B_ / 16, HID_ / 128, 1), 256, 0, stream>>>(
      euc16, 0, DIMF, p116, 0, HID_, nullptr, 0, 0, nullptr, 0, 0,
      h16, 0, HID_, B_, HID_, DIMF, p1b, HID_, 1 /*relu*/, 1.f);
  gemm16<false><<<dim3(B_ / 16, cdiv(NOUTP, 128), 1), 256, 0, stream>>>(
      h16, 0, HID_, p216, 0, NOUTP, nullptr, 0, 0, nullptr, 0, 0,
      out16, 0, NOUTP, B_, NOUTP, HID_, p2b, NOUT, 0, 1.f);
  gemm16<true><<<dim3(B_ / 16, NANS / 128, 1), 256, 0, stream>>>(
      out16, 0, NOUTP, gl16, 0, NOUTP, nullptr, 0, 0,
      outp, 0, NANS, nullptr, 0, 0, B_, NANS, NOUTP, nullptr, 0, 0, 1.f);
  log_softmax_rows<<<B_, 256, 0, stream>>>(outp, NANS);
}
